// FLARE_74912819577587
// MI455X (gfx1250) — compile-verified, matrix operand order checked by reading, not run on hardware
//
#include <hip/hip_runtime.h>
#include <hip/hip_bf16.h>
#include <math.h>

// ---------------------------------------------------------------------------
// FLARE fused pipeline for gfx1250 (MI455X), wave32, WMMA f16 (fp32 accum).
//
// Shapes: B=4, N=16384, DIM=512, H=8, D=64, S=64.  M = B*N = 65536.
//   Wxk = Wx . blockdiag(Wk), bias folded               (prep kernel)
//   k = x @ Wxk + bxk  -> [b][h][n][d] f16              (gemm MODE 0)
//   v = x @ Wxv + bxv  -> [b][h][d][n] f16 (transposed) (gemm MODE 2)
//   pass1: z[b,h,s,:] = softmax_n(G.k^T) @ v            (flash over N, 8 chunks)
//   pass2: y = softmax_s(k.G^T) @ z                     (local softmax, S=64)
//   out = y @ Wo + bo                                   (gemm MODE 1, f32 out)
// Softmax row-sums are computed with an extra WMMA against an all-ones
// B fragment; only row-max uses cross-lane shuffles.
// ---------------------------------------------------------------------------

typedef __attribute__((ext_vector_type(16))) _Float16 v16h;
typedef __attribute__((ext_vector_type(8)))  _Float16 v8h;
typedef __attribute__((ext_vector_type(2)))  __fp16   v2fp;   // cvt_pkrtz result
typedef __attribute__((ext_vector_type(8)))  float    v8f;

#define BB    4
#define NTOK  16384
#define DIMC  512
#define HH    8
#define DD    64
#define SS    64
#define MTOT  (BB * NTOK)       // 65536
#define NCHNK 8                 // pass-1 chunks per (b,h)
#define CHUNK (NTOK / NCHNK)    // 2048

__device__ __forceinline__ v8f wmma16(v16h a, v16h b, v8f c) {
    return __builtin_amdgcn_wmma_f32_16x16x32_f16(
        false, a, false, b, (short)0, c, false, false);
}

__device__ __forceinline__ v16h ones16() {
    v16h r;
#pragma unroll
    for (int i = 0; i < 16; ++i) r[i] = (_Float16)1.0f;
    return r;
}

__device__ __forceinline__ unsigned pk_f16(float a, float b) {
    v2fp t = __builtin_amdgcn_cvt_pkrtz(a, b);   // v_cvt_pk_rtz_f16_f32
    union { v2fp h; unsigned u; } c; c.h = t; return c.u;
}

// 16x32 f16 operand fragment from an LDS tile stored [row][pitch] row-major.
// A-operand: row = M index.  B-operand: tile stored [col][k]; row = N index.
// Per-lane K layout (ISA 7.12.2): base = 8*(lane>=16),
// K in {base..base+7, base+16..base+23}.
__device__ __forceinline__ v16h ld_frag(const _Float16* t, int row, int pitch,
                                        int koff, int lane) {
    const _Float16* p = t + row * pitch + koff + ((lane >> 4) << 3);
    v8h lo = *(const v8h*)(p);
    v8h hi = *(const v8h*)(p + 16);
    v16h r;
#pragma unroll
    for (int i = 0; i < 8; ++i) { r[i] = lo[i]; r[i + 8] = hi[i]; }
    return r;
}

__device__ __forceinline__ float rowmax16(float v) {
#pragma unroll
    for (int m = 1; m < 16; m <<= 1) v = fmaxf(v, __shfl_xor(v, m, 32));
    return v;
}

#define ZERO8F {0.f,0.f,0.f,0.f,0.f,0.f,0.f,0.f}

// ---------------------------------------------------------------------------
// Kernel 0a: fold Wk/Wv into Wx; transpose Wo; fold biases; -> f16.
// ---------------------------------------------------------------------------
__global__ void flare_prep_weights(const float* __restrict__ Wx,
                                   const float* __restrict__ bx,
                                   const float* __restrict__ Wk,
                                   const float* __restrict__ bk,
                                   const float* __restrict__ Wv,
                                   const float* __restrict__ bv,
                                   const float* __restrict__ Wo,
                                   _Float16* __restrict__ wxkT,
                                   _Float16* __restrict__ wxvT,
                                   _Float16* __restrict__ woT,
                                   float* __restrict__ bxk,
                                   float* __restrict__ bxv) {
    const int n = blockIdx.x;          // fused output column (h*64 + j)
    const int h = n >> 6, j = n & 63;
    for (int c = threadIdx.x; c < DIMC; c += blockDim.x) {
        float ak = 0.f, av = 0.f;
#pragma unroll 8
        for (int d = 0; d < DD; ++d) {
            float w = Wx[(size_t)c * DIMC + h * DD + d];
            ak += w * Wk[d * DD + j];
            av += w * Wv[d * DD + j];
        }
        wxkT[(size_t)n * DIMC + c] = (_Float16)ak;
        wxvT[(size_t)n * DIMC + c] = (_Float16)av;
        woT [(size_t)n * DIMC + c] = (_Float16)Wo[(size_t)c * DIMC + n];
    }
    if (threadIdx.x == 0) {
        float sk = bk[j], sv = bv[j];
        for (int d = 0; d < DD; ++d) {
            float b = bx[h * DD + d];
            sk += b * Wk[d * DD + j];
            sv += b * Wv[d * DD + j];
        }
        bxk[n] = sk; bxv[n] = sv;
    }
}

__global__ void flare_prep_g(const float* __restrict__ qg,
                             _Float16* __restrict__ gf) {
    int i = blockIdx.x * blockDim.x + threadIdx.x;
    if (i < HH * SS * DD) gf[i] = (_Float16)qg[i];
}

// ---------------------------------------------------------------------------
// Generic GEMM: C[M,512] = A[M,512] @ B^T + bias.  BT stored [col][k] f16.
// WG = 128 threads (4 waves); 128x64 tile; each wave: 32 rows x 64 cols,
// 8 WMMA per K-step (B fragments reused across the two row strips).
//   MODE 0: f16 out, k layout [b][h][n][d]
//   MODE 1: f32 out, row-major [m][512]
//   MODE 2: f16 out, transposed v layout [b][h][d][n]  (contiguous v8h store)
// ---------------------------------------------------------------------------
template <bool A_F32, int MODE>
__global__ void __launch_bounds__(128, 1)
flare_gemm(const void* __restrict__ Ain,
           const _Float16* __restrict__ BT,
           const float* __restrict__ bias,
           void* __restrict__ out) {
    __shared__ __align__(16) _Float16 at[128 * 32];
    __shared__ __align__(16) _Float16 bt[64 * 32];

    const int tid  = threadIdx.x;
    const int lane = tid & 31;
    const int wv   = tid >> 5;
    const int m0   = blockIdx.x * 128;
    const int c0   = blockIdx.y * 64;
    const int rb   = (lane >> 4) << 3;     // row base within 16x16 D tile

    v8f acc[2][4] = {{ZERO8F, ZERO8F, ZERO8F, ZERO8F},
                     {ZERO8F, ZERO8F, ZERO8F, ZERO8F}};

    for (int kk = 0; kk < DIMC; kk += 32) {
        // ---- stage A (128 x 32) tile ----
#pragma unroll
        for (int it = 0; it < 4; ++it) {
            int ci  = tid + it * 128;      // 512 chunks of 8 f16
            int row = ci >> 2;
            int kc  = (ci & 3) * 8;
            if (A_F32) {
                const float* s = (const float*)Ain +
                                 (size_t)(m0 + row) * DIMC + kk + kc;
                float4 f0 = *(const float4*)(s);
                float4 f1 = *(const float4*)(s + 4);
                uint4 pk;
                pk.x = pk_f16(f0.x, f0.y); pk.y = pk_f16(f0.z, f0.w);
                pk.z = pk_f16(f1.x, f1.y); pk.w = pk_f16(f1.z, f1.w);
                *(uint4*)&at[row * 32 + kc] = pk;
                if (kk + 32 < DIMC) __builtin_prefetch(s + 32, 0, 1);
            } else {
                const _Float16* s = (const _Float16*)Ain +
                                    (size_t)(m0 + row) * DIMC + kk + kc;
                *(v8h*)&at[row * 32 + kc] = *(const v8h*)s;
                if (kk + 32 < DIMC) __builtin_prefetch(s + 32, 0, 1);
            }
        }
        // ---- stage B (64 cols x 32 k) tile ----
#pragma unroll
        for (int it = 0; it < 2; ++it) {
            int ci  = tid + it * 128;      // 256 chunks of 8 f16
            int row = ci >> 2;
            int kc  = (ci & 3) * 8;
            const _Float16* sb = BT + (size_t)(c0 + row) * DIMC + kk + kc;
            *(v8h*)&bt[row * 32 + kc] = *(const v8h*)sb;
        }
        __syncthreads();

        // ---- 8 WMMA per wave per K-step ----
        v16h a0 = ld_frag(at, wv * 32 + (lane & 15), 32, 0, lane);
        v16h a1 = ld_frag(at, wv * 32 + 16 + (lane & 15), 32, 0, lane);
#pragma unroll
        for (int cs = 0; cs < 4; ++cs) {
            v16h b = ld_frag(bt, cs * 16 + (lane & 15), 32, 0, lane);
            acc[0][cs] = wmma16(a0, b, acc[0][cs]);
            acc[1][cs] = wmma16(a1, b, acc[1][cs]);
        }
        __syncthreads();
    }

    // ---- epilogue: bias + store ----
#pragma unroll
    for (int half = 0; half < 2; ++half)
#pragma unroll
        for (int cs = 0; cs < 4; ++cs) {
            int   c  = c0 + cs * 16 + (lane & 15);
            float bv = bias[c];
            int   mfirst = m0 + wv * 32 + half * 16 + rb;
            if (MODE == 2) {
                // v^T layout: [b][h][d][n] -- 8 consecutive n per lane
                int b = mfirst >> 14, n = mfirst & (NTOK - 1);
                int h = c >> 6,  d = c & 63;
                v8h pk;
#pragma unroll
                for (int j = 0; j < 8; ++j)
                    pk[j] = (_Float16)(acc[half][cs][j] + bv);
                *(v8h*)&((_Float16*)out)[((size_t)(b * HH + h) * DD + d) * NTOK + n] = pk;
            } else {
#pragma unroll
                for (int j = 0; j < 8; ++j) {
                    int   m   = mfirst + j;
                    float val = acc[half][cs][j] + bv;
                    if (MODE == 0) {
                        int b = m >> 14, n = m & (NTOK - 1);
                        int h = c >> 6,  d = c & 63;
                        ((_Float16*)out)[(((size_t)(b * HH + h) * NTOK) + n) * DD + d] =
                            (_Float16)val;
                    } else {
                        ((float*)out)[(size_t)m * DIMC + c] = val;
                    }
                }
            }
        }
}

// ---------------------------------------------------------------------------
// Pass 1: flash aggregation over tokens.  grid(B*H, NCHNK), 128 threads.
// Wave w owns s-rows [16w,16w+16).  Per 32-token step: scores = G.k^T (WMMA),
// online softmax over n (row-max via shuffles, row-sum via WMMA with ones),
// z += p.v^T (WMMA; v already stored transposed).  Writes (m,l,z) partials.
// ---------------------------------------------------------------------------
__global__ void __launch_bounds__(128, 1)
flare_pass1(const _Float16* __restrict__ kf,
            const _Float16* __restrict__ vtf,   // [b][h][d][n]
            const _Float16* __restrict__ gf,
            float* __restrict__ m_part,
            float* __restrict__ l_part,
            float* __restrict__ z_part) {
    __shared__ __align__(16) _Float16 gt[SS * DD];       // G[h]   [s][d]
    __shared__ __align__(16) _Float16 kt[32 * DD];       // k tile [n][d]
    __shared__ __align__(16) _Float16 vt[DD * 32];       // v^T    [d][n]
    __shared__ __align__(16) _Float16 pt[4 * 16 * 32];   // p per wave [s][n]

    const int bh    = blockIdx.x;
    const int chunk = blockIdx.y;
    const int h     = bh & (HH - 1);
    const int tid   = threadIdx.x;
    const int lane  = tid & 31;
    const int wv    = tid >> 5;
    const int rb    = (lane >> 4) << 3;
    const int sr0   = wv * 16;

    const _Float16* kbh = kf  + (size_t)bh * NTOK * DD;
    const _Float16* vbh = vtf + (size_t)bh * DD * NTOK;

    // stage G[h] once
#pragma unroll
    for (int it = 0; it < 4; ++it) {
        int ci = tid + it * 128;                  // 512 chunks of 8
        int r = ci >> 3, kc = (ci & 7) * 8;
        *(v8h*)&gt[r * DD + kc] =
            *(const v8h*)&gf[(size_t)h * SS * DD + r * DD + kc];
    }
    __syncthreads();

    const v16h vone = ones16();
    float mrow[8], lrow[8];
    v8f zacc[4] = {ZERO8F, ZERO8F, ZERO8F, ZERO8F};
#pragma unroll
    for (int j = 0; j < 8; ++j) { mrow[j] = -1e30f; lrow[j] = 0.f; }

    const int n_base = chunk * CHUNK;
    for (int st = 0; st < CHUNK / 32; ++st) {
        const int n0 = n_base + st * 32;
        // ---- stage k tile [32 n][64 d] ----
#pragma unroll
        for (int it = 0; it < 2; ++it) {
            int ci = tid + it * 128;              // 256 chunks of 8
            int r = ci >> 3, kc = (ci & 7) * 8;
            *(v8h*)&kt[r * DD + kc] =
                *(const v8h*)&kbh[(size_t)(n0 + r) * DD + kc];
        }
        // ---- stage v^T tile [64 d][32 n] (already transposed in memory) ----
#pragma unroll
        for (int it = 0; it < 2; ++it) {
            int ci = tid + it * 128;              // 256 chunks of 8
            int r = ci >> 2, kc = (ci & 3) * 8;
            *(v8h*)&vt[r * 32 + kc] =
                *(const v8h*)&vbh[(size_t)r * NTOK + n0 + kc];
        }
        __syncthreads();

        // ---- scores: [16 s x 32 n] in two 16-col tiles ----
        v8f s0 = ZERO8F, s1 = ZERO8F;
#pragma unroll
        for (int ds = 0; ds < 2; ++ds) {
            v16h ga = ld_frag(gt, sr0 + (lane & 15), DD, ds * 32, lane);
            v16h b0 = ld_frag(kt, 0 + (lane & 15), DD, ds * 32, lane);
            v16h b1 = ld_frag(kt, 16 + (lane & 15), DD, ds * 32, lane);
            s0 = wmma16(ga, b0, s0);
            s1 = wmma16(ga, b1, s1);
        }

        // ---- online softmax: row max (shuffles), write numerators ----
        _Float16* myp = pt + wv * (16 * 32);
#pragma unroll
        for (int j = 0; j < 8; ++j) {
            float mt   = rowmax16(fmaxf(s0[j], s1[j]));
            float mnew = fmaxf(mrow[j], mt);
            float scf  = __expf(mrow[j] - mnew);
            float e0   = __expf(s0[j] - mnew);
            float e1   = __expf(s1[j] - mnew);
            mrow[j] = mnew;
            lrow[j] *= scf;
#pragma unroll
            for (int dd = 0; dd < 4; ++dd) zacc[dd][j] *= scf;
            myp[(rb + j) * 32 + (lane & 15)]      = (_Float16)e0;
            myp[(rb + j) * 32 + 16 + (lane & 15)] = (_Float16)e1;
        }

        // ---- row sums via WMMA with ones; z += p @ v^T ----
        v16h ap = ld_frag(myp, lane & 15, 32, 0, lane);
        v8f ls = ZERO8F;
        ls = wmma16(ap, vone, ls);
#pragma unroll
        for (int j = 0; j < 8; ++j) lrow[j] += ls[j];
#pragma unroll
        for (int dd = 0; dd < 4; ++dd) {
            v16h bv = ld_frag(vt, dd * 16 + (lane & 15), 32, 0, lane);
            zacc[dd] = wmma16(ap, bv, zacc[dd]);
        }
        __syncthreads();
    }

    // ---- write chunk partials ----
    const int sbase = sr0 + rb;
    float* mp = m_part + (size_t)(bh * NCHNK + chunk) * SS;
    float* lp = l_part + (size_t)(bh * NCHNK + chunk) * SS;
    float* zp = z_part + (size_t)(bh * NCHNK + chunk) * SS * DD;
    if ((lane & 15) == 0) {
#pragma unroll
        for (int j = 0; j < 8; ++j) { mp[sbase + j] = mrow[j];
                                      lp[sbase + j] = lrow[j]; }
    }
#pragma unroll
    for (int dd = 0; dd < 4; ++dd)
#pragma unroll
        for (int j = 0; j < 8; ++j)
            zp[(size_t)(sbase + j) * DD + dd * 16 + (lane & 15)] = zacc[dd][j];
}

// ---------------------------------------------------------------------------
// Combine chunk partials -> z^T f16 [bh][d][s].  grid(B*H), 256 threads.
// ---------------------------------------------------------------------------
__global__ void flare_combine(const float* __restrict__ m_part,
                              const float* __restrict__ l_part,
                              const float* __restrict__ z_part,
                              _Float16* __restrict__ zT) {
    __shared__ float ml[SS], ll[SS];
    const int bh = blockIdx.x, tid = threadIdx.x;
    if (tid < SS) {
        float m = -1e30f;
        for (int c = 0; c < NCHNK; ++c)
            m = fmaxf(m, m_part[(size_t)(bh * NCHNK + c) * SS + tid]);
        float l = 0.f;
        for (int c = 0; c < NCHNK; ++c)
            l += __expf(m_part[(size_t)(bh * NCHNK + c) * SS + tid] - m) *
                 l_part[(size_t)(bh * NCHNK + c) * SS + tid];
        ml[tid] = m; ll[tid] = l;
    }
    __syncthreads();
    for (int idx = tid; idx < SS * DD; idx += blockDim.x) {
        int s = idx >> 6, d = idx & 63;
        float z = 0.f;
        for (int c = 0; c < NCHNK; ++c)
            z += __expf(m_part[(size_t)(bh * NCHNK + c) * SS + s] - ml[s]) *
                 z_part[((size_t)(bh * NCHNK + c) * SS + s) * DD + d];
        zT[(size_t)bh * DD * SS + d * SS + s] = (_Float16)(z / ll[s]);
    }
}

// ---------------------------------------------------------------------------
// Pass 2: y = softmax_s(k.G^T) @ z.  grid(M/64, H), 128 threads.
// Row-sum via WMMA-with-ones; normalization folded into the y epilogue
// using v_rcp_f32.
// ---------------------------------------------------------------------------
__global__ void __launch_bounds__(128, 1)
flare_pass2(const _Float16* __restrict__ kf,
            const _Float16* __restrict__ gf,
            const _Float16* __restrict__ zT,
            _Float16* __restrict__ yf) {
    __shared__ __align__(16) _Float16 gt[SS * DD];      // G[h]   [s][d]
    __shared__ __align__(16) _Float16 zt[DD * SS];      // z^T    [d][s]
    __shared__ __align__(16) _Float16 kt[64 * DD];      // k tile [n][d]
    __shared__ __align__(16) _Float16 pt[4 * 16 * SS];  // e per wave [n][s]

    const int m0   = blockIdx.x * 64;
    const int h    = blockIdx.y;
    const int b    = m0 >> 14;
    const int n0   = m0 & (NTOK - 1);
    const int bh   = b * HH + h;
    const int tid  = threadIdx.x;
    const int lane = tid & 31;
    const int wv   = tid >> 5;
    const int rb   = (lane >> 4) << 3;
    const int tr0  = wv * 16;

    const _Float16* gsrc = gf + (size_t)h * SS * DD;
    const _Float16* zsrc = zT + (size_t)bh * DD * SS;
    const _Float16* ksrc = kf + ((size_t)bh * NTOK + n0) * DD;
    for (int ci = tid; ci < 3 * 512; ci += 128) {      // 3 arrays x 512 chunks
        int arr = ci / 512, cc = ci % 512;
        int r = cc >> 3, kc = (cc & 7) * 8;
        if (arr == 0)      *(v8h*)&gt[r * DD + kc] = *(const v8h*)&gsrc[(size_t)r * DD + kc];
        else if (arr == 1) *(v8h*)&zt[r * SS + kc] = *(const v8h*)&zsrc[(size_t)r * SS + kc];
        else               *(v8h*)&kt[r * DD + kc] = *(const v8h*)&ksrc[(size_t)r * DD + kc];
    }
    __syncthreads();

    // ---- scores2: [16 tokens x 64 s] ----
    v8f sc[4] = {ZERO8F, ZERO8F, ZERO8F, ZERO8F};
#pragma unroll
    for (int ds = 0; ds < 2; ++ds) {
        v16h a = ld_frag(kt, tr0 + (lane & 15), DD, ds * 32, lane);
#pragma unroll
        for (int ss = 0; ss < 4; ++ss) {
            v16h bfr = ld_frag(gt, ss * 16 + (lane & 15), DD, ds * 32, lane);
            sc[ss] = wmma16(a, bfr, sc[ss]);
        }
    }

    // ---- softmax over s: row max via shuffles, numerators to LDS ----
    _Float16* myp = pt + wv * (16 * SS);
#pragma unroll
    for (int j = 0; j < 8; ++j) {
        float m = fmaxf(fmaxf(sc[0][j], sc[1][j]), fmaxf(sc[2][j], sc[3][j]));
        m = rowmax16(m);
#pragma unroll
        for (int ss = 0; ss < 4; ++ss)
            myp[(rb + j) * SS + ss * 16 + (lane & 15)] =
                (_Float16)__expf(sc[ss][j] - m);
    }

    // ---- row sums via WMMA-with-ones, y = e @ z^T, normalize, store ----
    const v16h vone = ones16();
    v16h ap[2];
#pragma unroll
    for (int st = 0; st < 2; ++st)
        ap[st] = ld_frag(myp, lane & 15, SS, st * 32, lane);

    v8f ls = ZERO8F;
    ls = wmma16(ap[0], vone, ls);
    ls = wmma16(ap[1], vone, ls);
    float linv[8];
#pragma unroll
    for (int j = 0; j < 8; ++j) linv[j] = __builtin_amdgcn_rcpf(ls[j]);

#pragma unroll
    for (int dd = 0; dd < 4; ++dd) {
        v8f y = ZERO8F;
#pragma unroll
        for (int st = 0; st < 2; ++st) {
            v16h bz = ld_frag(zt, dd * 16 + (lane & 15), SS, st * 32, lane);
            y = wmma16(ap[st], bz, y);
        }
#pragma unroll
        for (int j = 0; j < 8; ++j)
            yf[(size_t)(m0 + tr0 + rb + j) * DIMC + h * DD + dd * 16 + (lane & 15)] =
                (_Float16)(y[j] * linv[j]);
    }
}

// ---------------------------------------------------------------------------
// Launcher
// ---------------------------------------------------------------------------
extern "C" void kernel_launch(void* const* d_in, const int* in_sizes, int n_in,
                              void* d_out, int out_size, void* d_ws, size_t ws_size,
                              hipStream_t stream) {
    const float* x  = (const float*)d_in[0];
    const float* qg = (const float*)d_in[1];
    const float* Wx = (const float*)d_in[2];
    const float* bx = (const float*)d_in[3];
    const float* Wk = (const float*)d_in[4];
    const float* bk = (const float*)d_in[5];
    const float* Wv = (const float*)d_in[6];
    const float* bv = (const float*)d_in[7];
    const float* Wo = (const float*)d_in[8];
    const float* bo = (const float*)d_in[9];

    char* ws = (char*)d_ws;
    size_t o = 0;
    auto carve = [&](size_t bytes) { char* p = ws + o; o += (bytes + 255) & ~(size_t)255; return p; };

    _Float16* kf   = (_Float16*)carve((size_t)MTOT * DIMC * 2);   // 64 MB
    _Float16* vtf  = (_Float16*)carve((size_t)MTOT * DIMC * 2);   // 64 MB (v^T)
    _Float16* yf   = (_Float16*)carve((size_t)MTOT * DIMC * 2);   // 64 MB
    _Float16* wxkT = (_Float16*)carve((size_t)DIMC * DIMC * 2);
    _Float16* wxvT = (_Float16*)carve((size_t)DIMC * DIMC * 2);
    _Float16* woT  = (_Float16*)carve((size_t)DIMC * DIMC * 2);
    float*    bxk  = (float*)carve(DIMC * 4);
    float*    bxv  = (float*)carve(DIMC * 4);
    _Float16* gf   = (_Float16*)carve((size_t)HH * SS * DD * 2);
    float*    mpt  = (float*)carve((size_t)BB * HH * NCHNK * SS * 4);
    float*    lpt  = (float*)carve((size_t)BB * HH * NCHNK * SS * 4);
    float*    zpt  = (float*)carve((size_t)BB * HH * NCHNK * SS * DD * 4);
    _Float16* zTb  = (_Float16*)carve((size_t)BB * HH * DD * SS * 2);

    flare_prep_weights<<<DIMC, 128, 0, stream>>>(Wx, bx, Wk, bk, Wv, bv, Wo,
                                                 wxkT, wxvT, woT, bxk, bxv);
    flare_prep_g<<<(HH * SS * DD + 255) / 256, 256, 0, stream>>>(qg, gf);

    dim3 ggrid(MTOT / 128, DIMC / 64);
    flare_gemm<true, 0><<<ggrid, 128, 0, stream>>>(x, wxkT, bxk, kf);
    flare_gemm<true, 2><<<ggrid, 128, 0, stream>>>(x, wxvT, bxv, vtf);

    flare_pass1<<<dim3(BB * HH, NCHNK), 128, 0, stream>>>(kf, vtf, gf, mpt, lpt, zpt);
    flare_combine<<<BB * HH, 256, 0, stream>>>(mpt, lpt, zpt, zTb);
    flare_pass2<<<dim3(MTOT / 64, HH), 128, 0, stream>>>(kf, gf, zTb, yf);

    flare_gemm<false, 1><<<ggrid, 128, 0, stream>>>(yf, woT, bo, d_out);
}
